// InpaintGenerator_10926396801343
// MI455X (gfx1250) — compile-verified, matrix operand order checked by reading, not run on hardware
//
#include <hip/hip_runtime.h>
#include <hip/hip_bf16.h>
#include <math.h>

typedef __attribute__((ext_vector_type(16))) __bf16 v16bf;
typedef __attribute__((ext_vector_type(8)))  float  v8f;
typedef int vi4 __attribute__((vector_size(4 * sizeof(int))));  // matches builtin param

#define DEV __device__ __forceinline__

#if defined(__has_builtin)
#if __has_builtin(__builtin_amdgcn_global_load_async_to_lds_b128)
#define HAVE_ASYNC_LDS 1
#endif
#if __has_builtin(__builtin_amdgcn_s_wait_asynccnt)
#define HAVE_WAIT_ASYNC 1
#endif
#endif

DEV __bf16 f2bf(float f) {
  unsigned u = __builtin_bit_cast(unsigned, f);
  unsigned r = u + 0x7fffu + ((u >> 16) & 1u);   // round-to-nearest-even
  return __builtin_bit_cast(__bf16, (unsigned short)(r >> 16));
}

// Fragment = two contiguous 16B runs per lane: k = k0+8*h2+{0..7} and +16.
union Frag { v16bf v; uint4 q[2]; };

DEV v8f wmma_bf16(v16bf a, v16bf b, v8f c) {
  return __builtin_amdgcn_wmma_f32_16x16x32_bf16(false, a, false, b, (short)0, c,
                                                 false, false);
}

// 16-byte global -> LDS copy; async (ASYNCcnt) when the CDNA5 builtin exists.
DEV void copy16_g2l(const __bf16* g, __bf16* l) {
#if HAVE_ASYNC_LDS
  __builtin_amdgcn_global_load_async_to_lds_b128(
      (__attribute__((address_space(1))) vi4*)(g),
      (__attribute__((address_space(3))) vi4*)(l), 0, 0);
#else
  *(uint4*)l = *(const uint4*)g;
#endif
}
DEV void async_copy_wait() {
#if HAVE_ASYNC_LDS
#if HAVE_WAIT_ASYNC
  __builtin_amdgcn_s_wait_asynccnt(0);
#else
  asm volatile("s_wait_asynccnt 0" ::: "memory");
#endif
#endif
}

// ---------------------------------------------------------------- pack kernels
__global__ void k_pack_bf16(const float* __restrict__ s, __bf16* __restrict__ d, int n) {
  int i = blockIdx.x * blockDim.x + threadIdx.x;
  if (i < n) d[i] = f2bf(s[i]);
}

// NCHW fp32 -> NHWC bf16 (16,60*108,128)
__global__ void k_pack_nhwc(const float* __restrict__ s, __bf16* __restrict__ d, int n) {
  int i = blockIdx.x * blockDim.x + threadIdx.x;
  if (i >= n) return;
  int c = i & 127, pix = i >> 7;
  int img = pix / 6480, hw = pix % 6480;
  d[i] = f2bf(s[((long)img * 128 + c) * 6480 + hw]);
}

// OIHW 3x3 (128,128,3,3) -> [tap][oc][ic] bf16
__global__ void k_pack_w3(const float* __restrict__ s, __bf16* __restrict__ d) {
  int i = blockIdx.x * blockDim.x + threadIdx.x;
  if (i >= 128 * 128 * 9) return;
  int oc = i / (128 * 9), ic = (i / 9) % 128, t = i % 9;
  d[(t * 128 + oc) * 128 + ic] = f2bf(s[i]);
}

// --------------------------------------- 1x1 conv = GEMM (M=128,K=128,N=103680)
// X NHWC bf16; Wt [oc][ic] bf16; out NCHW fp32. grid.x = 6480 (exact), 8 waves.
__global__ __launch_bounds__(256) void k_gemm1x1_wmma(
    const __bf16* __restrict__ Xh, const __bf16* __restrict__ Wt,
    const float* __restrict__ bias, float* __restrict__ outF) {
  const int lane = threadIdx.x & 31, wave = threadIdx.x >> 5;
  const int m = lane & 15, h2 = lane >> 4;
  const int oc0 = wave * 16;
  const int pix = blockIdx.x * 16 + m;
  const __bf16* bsrc = Xh + ((long)pix << 7);
  const __bf16* asrc = Wt + (long)(oc0 + m) * 128;
  v8f acc = {};
#pragma unroll
  for (int k0 = 0; k0 < 128; k0 += 32) {
    Frag A, B;
    A.q[0] = *(const uint4*)(asrc + k0 + 8 * h2);
    A.q[1] = *(const uint4*)(asrc + k0 + 8 * h2 + 16);
    B.q[0] = *(const uint4*)(bsrc + k0 + 8 * h2);
    B.q[1] = *(const uint4*)(bsrc + k0 + 8 * h2 + 16);
    acc = wmma_bf16(A.v, B.v, acc);
  }
  int img = pix / 6480, hw = pix % 6480;
#pragma unroll
  for (int r = 0; r < 8; ++r) {
    int oc = oc0 + r + 8 * h2;
    outF[((long)img * 128 + oc) * 6480 + hw] = acc[r] + bias[oc];
  }
}

// ------------------- 3x3 conv (dil 1 or 2) via LDS-staged WMMA implicit GEMM
// X NHWC bf16; Wt [tap][oc][ic]; resid NCHW fp32; outF NCHW fp32; outBh NHWC bf16.
// grid.x = img(16) * y(60) * xchunk(7); block tile: 16 pixels x 128 oc.
#define TCOLS 20
__global__ __launch_bounds__(256) void k_conv3_wmma(
    const __bf16* __restrict__ Xh, const __bf16* __restrict__ Wt,
    const float* __restrict__ bias, const float* __restrict__ resid,
    float* __restrict__ outF, __bf16* __restrict__ outBh,
    int dil, int act) {
  __shared__ __bf16 tile[3 * TCOLS * 128];         // 15,360 B
  const int tid = threadIdx.x;
  const int lane = tid & 31, wave = tid >> 5;
  const int m = lane & 15, h2 = lane >> 4;
  int blk = blockIdx.x;
  int xc = blk % 7, y = (blk / 7) % 60, img = blk / (7 * 60);
  int x0 = xc * 16;
  // zero-fill tile (covers OOB halo)
  {
    uint4 zz = {0u, 0u, 0u, 0u};
    uint4* t4 = (uint4*)tile;                      // 960 16B chunks
    for (int i = tid; i < 960; i += 256) t4[i] = zz;
  }
  __syncthreads();
  // stage: rows y+(r-1)*dil, cols x0-2+c  (async global->LDS where available)
  for (int i = tid; i < 960; i += 256) {
    int part = i & 15;
    int c = (i >> 4) % TCOLS;
    int r = i / (16 * TCOLS);
    int yy = y + (r - 1) * dil;
    int xx = x0 - 2 + c;
    if ((unsigned)yy < 60u && (unsigned)xx < 108u) {
      const __bf16* g = Xh + (((long)img * 6480 + yy * 108 + xx) << 7) + part * 8;
      __bf16* l = tile + (r * TCOLS + c) * 128 + part * 8;
      copy16_g2l(g, l);
    }
  }
  async_copy_wait();
  __syncthreads();
  v8f acc = {};
  for (int tap = 0; tap < 9; ++tap) {
    int r = tap / 3;
    int kx = (tap % 3 - 1) * dil;
    const __bf16* bsrc = tile + (r * TCOLS + (m + 2 + kx)) * 128;
    const __bf16* asrc = Wt + ((long)tap * 128 + wave * 16 + m) * 128;
#pragma unroll
    for (int k0 = 0; k0 < 128; k0 += 32) {
      Frag A, B;
      A.q[0] = *(const uint4*)(asrc + k0 + 8 * h2);
      A.q[1] = *(const uint4*)(asrc + k0 + 8 * h2 + 16);
      B.q[0] = *(const uint4*)(bsrc + k0 + 8 * h2);
      B.q[1] = *(const uint4*)(bsrc + k0 + 8 * h2 + 16);
      acc = wmma_bf16(A.v, B.v, acc);
    }
  }
  int x = x0 + m;
  if (x >= 108) return;
  int hw = y * 108 + x;
#pragma unroll
  for (int r = 0; r < 8; ++r) {
    int oc = wave * 16 + r + 8 * h2;
    float v = acc[r] + bias[oc];
    if (act) v = (v >= 0.f) ? v : 0.2f * v;        // LeakyReLU(0.2)
    long oF = ((long)img * 128 + oc) * 6480 + hw;
    if (resid) v += resid[oF];
    if (outF) outF[oF] = v;
    if (outBh) outBh[((long)img * 6480 + hw) * 128 + oc] = f2bf(v);
  }
}

// -------------------------------------------------- depthwise 3x3 add into v_all
__global__ void k_dwconv_add(const float* __restrict__ x, const float* __restrict__ w,
                             const float* __restrict__ b, float* __restrict__ v, int n) {
  int i = blockIdx.x * blockDim.x + threadIdx.x;
  if (i >= n) return;
  int hw = i % 6480, c = (i / 6480) % 128, img = i / (6480 * 128);
  int py = hw / 108, px = hw % 108;
  float acc = b[c];
  const float* xb = x + ((long)img * 128 + c) * 6480;
#pragma unroll
  for (int t = 0; t < 9; ++t) {
    int iy = py + t / 3 - 1, ix = px + t % 3 - 1;
    if ((unsigned)iy < 60u && (unsigned)ix < 108u) acc += w[c * 9 + t] * xb[iy * 108 + ix];
  }
  v[i] += acc;
}

// --------------------- fused depth encoder: two stride-2 3x3 convs + ReLUs
__global__ void k_depth_enc(const float* __restrict__ dm, const float* __restrict__ wd1,
                            const float* __restrict__ bd1, const float* __restrict__ wd2,
                            const float* __restrict__ bd2, float* __restrict__ dep) {
  int i = blockIdx.x * blockDim.x + threadIdx.x;
  if (i >= 16 * 6480) return;
  int hw = i % 6480, img = i / 6480;
  int y = hw / 108, x = hw % 108;
  const float* dmb = dm + (long)img * 240 * 432;
  float acc = bd2[0];
  for (int p = 0; p < 9; ++p) {
    int Y = 2 * y - 1 + p / 3, X = 2 * x - 1 + p % 3;
    if ((unsigned)Y >= 120u || (unsigned)X >= 216u) continue;
    float s[9];
#pragma unroll
    for (int t = 0; t < 9; ++t) {
      int yy = 2 * Y - 1 + t / 3, xx = 2 * X - 1 + t % 3;
      s[t] = ((unsigned)yy < 240u && (unsigned)xx < 432u) ? dmb[yy * 432 + xx] : 0.f;
    }
    for (int c = 0; c < 128; ++c) {
      float a = bd1[c];
#pragma unroll
      for (int t = 0; t < 9; ++t) a += s[t] * wd1[c * 9 + t];
      a = a > 0.f ? a : 0.f;
      acc += wd2[c * 9 + p] * a;
    }
  }
  dep[i] = acc > 0.f ? acc : 0.f;
}

// ---------------- per key-patch stats: mask, sigmoid(min/max/lower-median(dep))
__global__ __launch_bounds__(256) void k_patch_stats(
    const float* __restrict__ mbuf, const float* __restrict__ dep,
    int* __restrict__ mask, float* __restrict__ smed, float* __restrict__ smax,
    float* __restrict__ smin, int t, int ph, int pw, int oh, int ow) {
  __shared__ float sd[6480];
  __shared__ float red[256];
  int L = t * oh * ow;
  int bl = blockIdx.x, bb = bl / L, l = bl % L;
  int tt = l / (oh * ow), oy = (l / ow) % oh, ox = l % ow;
  long base = (long)(bb * t + tt) * 6480;
  int n = ph * pw, tid = threadIdx.x;
  float msum = 0.f, mn = 3.4e38f, mx = -3.4e38f;
  for (int i = tid; i < n; i += 256) {
    int hw = (oy * ph + i / pw) * 108 + ox * pw + i % pw;
    float dv = dep[base + hw];
    sd[i] = dv;
    mn = fminf(mn, dv); mx = fmaxf(mx, dv);
    msum += mbuf[base + hw];
  }
  red[tid] = msum; __syncthreads();
  for (int s2 = 128; s2 > 0; s2 >>= 1) { if (tid < s2) red[tid] += red[tid + s2]; __syncthreads(); }
  float sumM = red[0]; __syncthreads();
  red[tid] = mn; __syncthreads();
  for (int s2 = 128; s2 > 0; s2 >>= 1) { if (tid < s2) red[tid] = fminf(red[tid], red[tid + s2]); __syncthreads(); }
  float vmn = red[0]; __syncthreads();
  red[tid] = mx; __syncthreads();
  for (int s2 = 128; s2 > 0; s2 >>= 1) { if (tid < s2) red[tid] = fmaxf(red[tid], red[tid + s2]); __syncthreads(); }
  float vmx = red[0]; __syncthreads();
  int target = (n - 1) / 2;                         // torch-style lower median
  for (int i = tid; i < n; i += 256) {
    float v = sd[i]; int r = 0;
    for (int j = 0; j < n; ++j) {
      float u = sd[j];
      r += (u < v) || (u == v && j < i);
    }
    if (r == target) smed[bl] = 1.f / (1.f + expf(-v));
  }
  if (tid == 0) {
    mask[bl] = (sumM / (float)n) > 0.5f;
    smax[bl] = 1.f / (1.f + expf(-vmx));
    smin[bl] = 1.f / (1.f + expf(-vmn));
  }
}

// -------------- patch partition: Q/K row-major (b,L,D); V transposed (b,D,Lpad)
__global__ void k_partition(const float* __restrict__ src, __bf16* __restrict__ dst,
                            int t, int cc0, int ph, int pw, int oh, int ow) {
  int L = t * oh * ow, D = 32 * ph * pw;
  long total = (long)2 * L * D;
  long i = (long)blockIdx.x * blockDim.x + threadIdx.x;
  if (i >= total) return;
  int dd = (int)(i % D); long rest = i / D; int l = (int)(rest % L); int bb = (int)(rest / L);
  int tt = l / (oh * ow), oy = (l / ow) % oh, ox = l % ow;
  int cc = dd / (ph * pw), pp = dd % (ph * pw);
  long s = ((long)(bb * t + tt) * 128 + cc0 + cc) * 6480 + (oy * ph + pp / pw) * 108 + ox * pw + pp % pw;
  dst[i] = f2bf(src[s]);
}

__global__ void k_partition_vT(const float* __restrict__ src, __bf16* __restrict__ dst,
                               int t, int cc0, int ph, int pw, int oh, int ow) {
  int L = t * oh * ow, D = 32 * ph * pw, Lpad = (L + 31) & ~31;
  long total = (long)2 * D * Lpad;
  long i = (long)blockIdx.x * blockDim.x + threadIdx.x;
  if (i >= total) return;
  int l = (int)(i % Lpad); long rest = i / Lpad; int dd = (int)(rest % D); int bb = (int)(rest / D);
  if (l >= L) { dst[i] = f2bf(0.f); return; }       // zero K-tail for clean WMMA
  int tt = l / (oh * ow), oy = (l / ow) % oh, ox = l % ow;
  int cc = dd / (ph * pw), pp = dd % (ph * pw);
  long s = ((long)(bb * t + tt) * 128 + cc0 + cc) * 6480 + (oy * ph + pp / pw) * 108 + ox * pw + pp % pw;
  dst[i] = f2bf(src[s]);
}

// (b,L,D) fp32 attention result -> NHWC bf16 att slice, with /3 mean
__global__ void k_unpartition(const float* __restrict__ Y, __bf16* __restrict__ atth,
                              int t, int cc0, int ph, int pw, int oh, int ow) {
  int L = t * oh * ow, D = 32 * ph * pw;
  long total = (long)2 * L * D;
  long i = (long)blockIdx.x * blockDim.x + threadIdx.x;
  if (i >= total) return;
  int dd = (int)(i % D); long rest = i / D; int l = (int)(rest % L); int bb = (int)(rest / L);
  int tt = l / (oh * ow), oy = (l / ow) % oh, ox = l % ow;
  int cc = dd / (ph * pw), pp = dd % (ph * pw);
  long hw = (long)(oy * ph + pp / pw) * 108 + ox * pw + pp % pw;
  atth[(((long)(bb * t + tt) * 6480) + hw) * 128 + cc0 + cc] = f2bf(Y[i] * (1.f / 3.f));
}

// ------------------------------------ S = Q K^T * rsqrt(D), WMMA, computed ONCE
__global__ __launch_bounds__(256) void k_attn_qk(const __bf16* __restrict__ Q,
    const __bf16* __restrict__ K, float* __restrict__ S, int L, int D, float scale) {
  int lane = threadIdx.x & 31, wave = threadIdx.x >> 5;
  int m = lane & 15, h2 = lane >> 4;
  int bb = blockIdx.z;
  int row = blockIdx.x * 16 + m;
  int col = blockIdx.y * 128 + wave * 16 + m;
  bool av = row < L, bv = col < L;
  const __bf16* Qb = Q + ((long)bb * L + (av ? row : 0)) * D;
  const __bf16* Kb = K + ((long)bb * L + (bv ? col : 0)) * D;
  v8f acc = {};
  uint4 zz = {0u, 0u, 0u, 0u};
  for (int k0 = 0; k0 < D; k0 += 32) {
    Frag A, B;
    if (av) { A.q[0] = *(const uint4*)(Qb + k0 + 8 * h2); A.q[1] = *(const uint4*)(Qb + k0 + 8 * h2 + 16); }
    else    { A.q[0] = zz; A.q[1] = zz; }
    if (bv) { B.q[0] = *(const uint4*)(Kb + k0 + 8 * h2); B.q[1] = *(const uint4*)(Kb + k0 + 8 * h2 + 16); }
    else    { B.q[0] = zz; B.q[1] = zz; }
    acc = wmma_bf16(A.v, B.v, acc);
  }
#pragma unroll
  for (int r = 0; r < 8; ++r) {
    int rr = blockIdx.x * 16 + r + 8 * h2;
    if (rr < L && col < L) S[((long)bb * L + rr) * L + col] = acc[r] * scale;
  }
}

// ---------- masked, depth-scaled softmax -> P bf16 rows padded to Lpad (zeros)
__global__ __launch_bounds__(128) void k_softmax(const float* __restrict__ S,
    const float* __restrict__ sig, const int* __restrict__ mask,
    __bf16* __restrict__ P, int L, int Lpad) {
  __shared__ float red[128];
  int row = blockIdx.x, bb = blockIdx.y, tid = threadIdx.x;
  const float* sr = S + ((long)bb * L + row) * L;
  __bf16* pr = P + ((long)bb * L + row) * Lpad;
  float mx = -3.4e38f;
  for (int k = tid; k < L; k += 128) {
    float s = mask[bb * L + k] ? -3.4e38f : sr[k] * sig[bb * L + k];
    mx = fmaxf(mx, s);
  }
  red[tid] = mx; __syncthreads();
  for (int s2 = 64; s2 > 0; s2 >>= 1) { if (tid < s2) red[tid] = fmaxf(red[tid], red[tid + s2]); __syncthreads(); }
  mx = red[0]; __syncthreads();
  float sum = 0.f;
  for (int k = tid; k < L; k += 128) {
    float s = mask[bb * L + k] ? -3.4e38f : sr[k] * sig[bb * L + k];
    sum += (mx > -3.3e38f) ? expf(s - mx) : 0.f;
  }
  red[tid] = sum; __syncthreads();
  for (int s2 = 64; s2 > 0; s2 >>= 1) { if (tid < s2) red[tid] += red[tid + s2]; __syncthreads(); }
  float inv = red[0] > 0.f ? 1.f / red[0] : 0.f;
  for (int k = tid; k < L; k += 128) {
    float s = mask[bb * L + k] ? -3.4e38f : sr[k] * sig[bb * L + k];
    float e = (mx > -3.3e38f) ? expf(s - mx) * inv : 0.f;
    pr[k] = f2bf(e);
  }
  for (int k = L + tid; k < Lpad; k += 128) pr[k] = f2bf(0.f);
}

// ------------- Y (+)= P V, WMMA; P (b,L,Lpad), Vt (b,D,Lpad) -> pure b128 loads
__global__ __launch_bounds__(256) void k_attn_pv(const __bf16* __restrict__ P,
    const __bf16* __restrict__ Vt, float* __restrict__ Y, int L, int Lpad, int D,
    int accum) {
  int lane = threadIdx.x & 31, wave = threadIdx.x >> 5;
  int m = lane & 15, h2 = lane >> 4;
  int bb = blockIdx.z;
  int row = blockIdx.y * 16 + m;
  int col = blockIdx.x * 128 + wave * 16 + m;
  bool av = row < L, cv = col < D;
  const __bf16* Pb = P + ((long)bb * L + (av ? row : 0)) * Lpad;
  const __bf16* Vb = Vt + ((long)bb * D + (cv ? col : 0)) * Lpad;
  v8f acc = {};
  uint4 zz = {0u, 0u, 0u, 0u};
  for (int k0 = 0; k0 < Lpad; k0 += 32) {
    Frag A, B;
    if (av) { A.q[0] = *(const uint4*)(Pb + k0 + 8 * h2); A.q[1] = *(const uint4*)(Pb + k0 + 8 * h2 + 16); }
    else    { A.q[0] = zz; A.q[1] = zz; }
    B.q[0] = *(const uint4*)(Vb + k0 + 8 * h2);
    B.q[1] = *(const uint4*)(Vb + k0 + 8 * h2 + 16);
    acc = wmma_bf16(A.v, B.v, acc);
  }
#pragma unroll
  for (int r = 0; r < 8; ++r) {
    int rr = blockIdx.y * 16 + r + 8 * h2;
    if (rr < L && cv) {
      long o = ((long)bb * L + rr) * D + col;
      Y[o] = accum ? Y[o] + acc[r] : acc[r];
    }
  }
}

// =============================================================================
extern "C" void kernel_launch(void* const* d_in, const int* in_sizes, int n_in,
                              void* d_out, int out_size, void* d_ws, size_t ws_size,
                              hipStream_t stream) {
  (void)in_sizes; (void)n_in; (void)out_size; (void)ws_size;
  const float* x    = (const float*)d_in[0];
  const float* mbuf = (const float*)d_in[1];
  const float* dm   = (const float*)d_in[2];
  const float* wq   = (const float*)d_in[3];  const float* bq   = (const float*)d_in[4];
  const float* wk   = (const float*)d_in[5];  const float* bk   = (const float*)d_in[6];
  const float* wv   = (const float*)d_in[7];  const float* bv   = (const float*)d_in[8];
  const float* wvle = (const float*)d_in[9];  const float* bvle = (const float*)d_in[10];
  const float* wout = (const float*)d_in[11]; const float* bout = (const float*)d_in[12];
  const float* wd1  = (const float*)d_in[13]; const float* bd1  = (const float*)d_in[14];
  const float* wd2  = (const float*)d_in[15]; const float* bd2  = (const float*)d_in[16];
  const float* wf1  = (const float*)d_in[17]; const float* bf1  = (const float*)d_in[18];
  const float* wf2  = (const float*)d_in[19]; const float* bf2  = (const float*)d_in[20];
  float* out = (float*)d_out;

  const int NX = 16 * 128 * 6480;            // 13,271,040
  const long LD   = (long)2 * 8 * 32 * 6480; // b*L*D (constant) = 3,317,760
  const long LDVT = (long)2 * 207360 * 32;   // max b*D*Lpad = 13,271,040
  const long SMAX = (long)2 * 1152 * 1152;

  size_t off = 0;
  char* base = (char*)d_ws;
  auto alloc = [&](size_t nbytes) { off = (off + 255) & ~(size_t)255; size_t o = off; off += nbytes; return o; };
  __bf16* xbh   = (__bf16*)(base + alloc((size_t)NX * 2));   // x NHWC bf16
  __bf16* wqb   = (__bf16*)(base + alloc(16384 * 2));
  __bf16* wkb   = (__bf16*)(base + alloc(16384 * 2));
  __bf16* wvb   = (__bf16*)(base + alloc(16384 * 2));
  __bf16* woutb = (__bf16*)(base + alloc(147456 * 2));
  __bf16* wf1b  = (__bf16*)(base + alloc(147456 * 2));
  __bf16* wf2b  = (__bf16*)(base + alloc(147456 * 2));
  float*  qA    = (float*)(base + alloc((size_t)NX * 4));    // NCHW fp32
  float*  kA    = (float*)(base + alloc((size_t)NX * 4));
  float*  vA    = (float*)(base + alloc((size_t)NX * 4));
  float*  dep   = (float*)(base + alloc((size_t)16 * 6480 * 4));
  int*    maskA = (int*)(base + alloc(2304 * 4));
  float*  smed  = (float*)(base + alloc(2304 * 4));
  float*  smax  = (float*)(base + alloc(2304 * 4));
  float*  smin  = (float*)(base + alloc(2304 * 4));
  __bf16* Qp    = (__bf16*)(base + alloc((size_t)LD * 2));
  __bf16* Kp    = (__bf16*)(base + alloc((size_t)LD * 2));
  __bf16* Vtp   = (__bf16*)(base + alloc((size_t)LDVT * 2));
  float*  Sb    = (float*)(base + alloc((size_t)SMAX * 4));
  __bf16* Pb    = (__bf16*)(base + alloc((size_t)SMAX * 2));
  float*  Yacc  = (float*)(base + alloc((size_t)LD * 4));
  __bf16* attbh = (__bf16*)(base + alloc((size_t)NX * 2));   // NHWC
  float*  x2f   = (float*)(base + alloc((size_t)NX * 4));    // NCHW
  __bf16* x2bh  = (__bf16*)(base + alloc((size_t)NX * 2));   // NHWC
  __bf16* ff1bh = (__bf16*)(base + alloc((size_t)NX * 2));   // NHWC

  // ---- pack activations + weights to bf16 (x -> NHWC for b128 GEMM feeds)
  k_pack_nhwc<<<(NX + 255) / 256, 256, 0, stream>>>(x, xbh, NX);
  k_pack_bf16<<<64, 256, 0, stream>>>(wq, wqb, 16384);
  k_pack_bf16<<<64, 256, 0, stream>>>(wk, wkb, 16384);
  k_pack_bf16<<<64, 256, 0, stream>>>(wv, wvb, 16384);
  k_pack_w3<<<(147456 + 255) / 256, 256, 0, stream>>>(wout, woutb);
  k_pack_w3<<<(147456 + 255) / 256, 256, 0, stream>>>(wf1, wf1b);
  k_pack_w3<<<(147456 + 255) / 256, 256, 0, stream>>>(wf2, wf2b);

  // ---- q/k/v 1x1 projections as WMMA GEMMs
  k_gemm1x1_wmma<<<6480, 256, 0, stream>>>(xbh, wqb, bq, qA);
  k_gemm1x1_wmma<<<6480, 256, 0, stream>>>(xbh, wkb, bk, kA);
  k_gemm1x1_wmma<<<6480, 256, 0, stream>>>(xbh, wvb, bv, vA);
  k_dwconv_add<<<(NX + 255) / 256, 256, 0, stream>>>(x, wvle, bvle, vA, NX);
  k_depth_enc<<<(16 * 6480 + 255) / 256, 256, 0, stream>>>(dm, wd1, bd1, wd2, bd2, dep);

  // ---- per-scale attention
  const int PH[4] = {60, 20, 10, 5}, PW[4] = {108, 36, 18, 9};
  for (int i = 0; i < 4; ++i) {
    int ph = PH[i], pw = PW[i], oh = 60 / ph, ow = 108 / pw;
    int L = 8 * oh * ow, D = 32 * ph * pw, cc0 = i * 32;
    int Lpad = (L + 31) & ~31;
    k_patch_stats<<<2 * L, 256, 0, stream>>>(mbuf, dep, maskA, smed, smax, smin, 8, ph, pw, oh, ow);
    long tot = (long)2 * L * D;
    int gpart = (int)((tot + 255) / 256);
    k_partition<<<gpart, 256, 0, stream>>>(qA, Qp, 8, cc0, ph, pw, oh, ow);
    k_partition<<<gpart, 256, 0, stream>>>(kA, Kp, 8, cc0, ph, pw, oh, ow);
    long totv = (long)2 * D * Lpad;
    k_partition_vT<<<(int)((totv + 255) / 256), 256, 0, stream>>>(vA, Vtp, 8, cc0, ph, pw, oh, ow);
    dim3 gqk((L + 15) / 16, (L + 127) / 128, 2);
    k_attn_qk<<<gqk, 256, 0, stream>>>(Qp, Kp, Sb, L, D, 1.0f / sqrtf((float)D));
    const float* sigs[3] = {smed, smax, smin};
    for (int vv = 0; vv < 3; ++vv) {
      k_softmax<<<dim3(L, 2), 128, 0, stream>>>(Sb, sigs[vv], maskA, Pb, L, Lpad);
      dim3 gpv((D + 127) / 128, (L + 15) / 16, 2);
      k_attn_pv<<<gpv, 256, 0, stream>>>(Pb, Vtp, Yacc, L, Lpad, D, vv);
    }
    k_unpartition<<<gpart, 256, 0, stream>>>(Yacc, attbh, 8, cc0, ph, pw, oh, ow);
  }

  // ---- output conv + residual, FFN (dilated conv) + residual, all WMMA+LDS
  k_conv3_wmma<<<16 * 60 * 7, 256, 0, stream>>>(attbh, woutb, bout, x, x2f, x2bh, 1, 1);
  k_conv3_wmma<<<16 * 60 * 7, 256, 0, stream>>>(x2bh, wf1b, bf1, nullptr, nullptr, ff1bh, 2, 1);
  k_conv3_wmma<<<16 * 60 * 7, 256, 0, stream>>>(ff1bh, wf2b, bf2, x2f, out, nullptr, 1, 1);
}
